// ConditionalWaveNet_63952063037811
// MI455X (gfx1250) — compile-verified, matrix operand order checked by reading, not run on hardware
//
#include <hip/hip_runtime.h>
#include <hip/hip_bf16.h>
#include <math.h>

// ---------------------------------------------------------------------------
// ConditionalWaveNet on MI455X (gfx1250, wave32, WMMA).
// All channel contractions run on v_wmma_f32_16x16x32_bf16 (bf16 in, fp32 acc).
// Residual / skip / gating / softmax stay fp32.
// Workspace layout (needs ws_size >= ~104 MB):
//   [0, L)        l ping   (B*64*T fp32, L = 33.5 MB)
//   [L, 2L)       l pong
//   [2L, 3L)      skip accumulator
//   [3L, +9*256K) per-block packed weights (bf16 A-fragments + fp32 gcond)
//   then          packed pre/out head weights
// ---------------------------------------------------------------------------

typedef __attribute__((ext_vector_type(16))) __bf16 v16bf;
typedef __attribute__((ext_vector_type(8)))  float  v8f;

#define WMMA_BF16(A, B, C) \
  __builtin_amdgcn_wmma_f32_16x16x32_bf16(false, (A), false, (B), (short)0, (C), false, false)

__device__ __forceinline__ unsigned short f2bf(float f) {
  union { float f; unsigned u; } x; x.f = f;
  unsigned u = x.u;
  u = u + 0x7FFFu + ((u >> 16) & 1u);   // round-to-nearest-even
  return (unsigned short)(u >> 16);
}

// 16-bit matrix fragment K-mapping (ISA 7.12.2):
// lane-group g (lane/16), half j in [0,16): K = (j<8 ? j : j+8) + 8*g
__device__ __forceinline__ int kmap(int g, int j) { return ((j < 8) ? j : j + 8) + (g << 3); }

// ---------------------------------------------------------------------------
// Pack one residual block's weights into bf16 A-fragment order + gcond = h@G^T
// frag layout (unsigned short units):
//   [0,          aSize)  A tanh tap0       aSize = F*64
//   [aSize,     2*aSize) A tanh tap1
//   [2*aSize,   3*aSize) A gate tap0
//   [3*aSize,   4*aSize) A gate tap1
//   [4*aSize, 4*aSize+64F) scale A-fragments (64 x F, k-tiled by 32)
// ---------------------------------------------------------------------------
__global__ __launch_bounds__(256) void pack_block(
    const float* __restrict__ Wt, const float* __restrict__ Wg,
    const float* __restrict__ Sw, const float* __restrict__ Gt,
    const float* __restrict__ Gg, const float* __restrict__ h,
    unsigned short* __restrict__ frag, float* __restrict__ gcT,
    float* __restrict__ gcG, int F)
{
  const int aSize = F * 64;
  const int nA = aSize, nS = 64 * F, nG = 8 * F;
  const int total = nA + nS + nG;
  for (int e = blockIdx.x * blockDim.x + threadIdx.x; e < total;
       e += gridDim.x * blockDim.x) {
    if (e < nA) {
      int j = e & 15, ln = (e >> 4) & 31, kt = (e >> 9) & 1, rt = e >> 10;
      int f = rt * 16 + (ln & 15);
      int c = kt * 32 + kmap(ln >> 4, j);
      int wi = f * 128 + c * 2;          // W[f][c][tap], shape [F,64,2]
      frag[e]             = f2bf(Wt[wi + 0]);
      frag[aSize + e]     = f2bf(Wt[wi + 1]);
      frag[2 * aSize + e] = f2bf(Wg[wi + 0]);
      frag[3 * aSize + e] = f2bf(Wg[wi + 1]);
    } else if (e < nA + nS) {
      int e2 = e - nA;
      int j = e2 & 15, ln = (e2 >> 4) & 31;
      int kTiles = F >> 5;
      int kt = (e2 >> 9) % kTiles, rt = (e2 >> 9) / kTiles;
      int o = rt * 16 + (ln & 15);
      int f = kt * 32 + kmap(ln >> 4, j);
      frag[4 * aSize + e2] = f2bf(Sw[o * F + f]);   // scale_W [64,F]
    } else {
      int e2 = e - nA - nS;
      int bb = e2 / F, f = e2 % F;
      float at = 0.f, ag = 0.f;
      for (int g = 0; g < 100; ++g) {
        float hv = h[bb * 100 + g];
        at += hv * Gt[f * 100 + g];
        ag += hv * Gg[f * 100 + g];
      }
      gcT[bb * F + f] = at;
      gcG[bb * F + f] = ag;
    }
  }
}

// pack pre_W [128,64] (2 k-tiles) and out_W [256,128] (4 k-tiles)
__global__ __launch_bounds__(256) void pack_head(
    const float* __restrict__ preW, const float* __restrict__ outW,
    unsigned short* __restrict__ preF, unsigned short* __restrict__ outF)
{
  const int nPre = 128 * 64, nOut = 256 * 128;
  for (int e = blockIdx.x * blockDim.x + threadIdx.x; e < nPre + nOut;
       e += gridDim.x * blockDim.x) {
    if (e < nPre) {
      int j = e & 15, ln = (e >> 4) & 31, kt = (e >> 9) & 1, rt = e >> 10;
      int o = rt * 16 + (ln & 15);
      int c = kt * 32 + kmap(ln >> 4, j);
      preF[e] = f2bf(preW[o * 64 + c]);
    } else {
      int e2 = e - nPre;
      int j = e2 & 15, ln = (e2 >> 4) & 31, kt = (e2 >> 9) & 3, rt = e2 >> 11;
      int o = rt * 16 + (ln & 15);
      int c = kt * 32 + kmap(ln >> 4, j);
      outF[e2] = f2bf(outW[o * 128 + c]);
    }
  }
}

// initial causal conv (64x1 k=2 d=1) + zero skip
__global__ __launch_bounds__(256) void wavenet_init(
    const float* __restrict__ data, const float* __restrict__ cW,
    const float* __restrict__ cb, float* __restrict__ l0,
    float* __restrict__ skip, int T)
{
  long i = (long)blockIdx.x * blockDim.x + threadIdx.x;
  long N = (long)8 * 64 * T;
  if (i >= N) return;
  int t = (int)(i % T);
  int c = (int)((i / T) % 64);
  int b = (int)(i / ((long)64 * T));
  float xp = (t > 0) ? data[(long)b * T + t - 1] : 0.f;
  float xc = data[(long)b * T + t];
  l0[i] = cW[c * 2] * xp + cW[c * 2 + 1] * xc + cb[c];
  skip[i] = 0.f;
}

// ---------------------------------------------------------------------------
// Fused residual block: dilated conv (tanh+gate) -> +global cond -> +local
// cond -> gate -> 1x1 scale -> residual + skip.  One workgroup = 32 timesteps.
// ---------------------------------------------------------------------------
__global__ __launch_bounds__(256) void wavenet_block(
    const float* __restrict__ l_in, float* __restrict__ l_out,
    float* __restrict__ skip, const float* __restrict__ y,
    const unsigned short* __restrict__ frag,
    const float* __restrict__ gcT, const float* __restrict__ gcG,
    const float* __restrict__ dil_tb, const float* __restrict__ loc_tb,
    const float* __restrict__ dil_gb, const float* __restrict__ loc_gb,
    const float* __restrict__ locTW, const float* __restrict__ locGW,
    const float* __restrict__ scale_b, int F, int d, int T)
{
  __shared__ float sLc[64][32];                       // l current tap (coalesced stage)
  __shared__ float sLp[64][32];                       // l dilated tap
  __shared__ __align__(32) unsigned short sBc[2][2][512];  // B-frag cur  [kt][nt]
  __shared__ __align__(32) unsigned short sBp[2][2][512];  // B-frag prev
  __shared__ __align__(32) unsigned short sM[4][2][512];   // gated m frags, K=F
  __shared__ float sYc[32], sYp[32];

  const int b = blockIdx.y;
  const int t0 = blockIdx.x * 32;
  const int tid = threadIdx.x;
  const int lane = tid & 31, wv = tid >> 5;
  const float* lb = l_in + (size_t)b * 64 * T;

  // phase 0: coalesced stage of l tiles
  for (int e = tid; e < 2048; e += 256) {
    int c = e >> 5, tc = e & 31;
    sLc[c][tc] = lb[(size_t)c * T + t0 + tc];
    int tp = t0 + tc - d;
    sLp[c][tc] = (tp >= 0) ? lb[(size_t)c * T + tp] : 0.f;
  }
  if (tid < 32) {
    sYc[tid] = y[(size_t)b * T + t0 + tid];
    int tp = t0 + tid - d;
    sYp[tid] = (tp >= 0) ? y[(size_t)b * T + tp] : 0.f;
  }
  __syncthreads();

  // phase 1: repack to bf16 B-fragments
  for (int e = tid; e < 2048; e += 256) {
    int j = e & 15, ln = (e >> 4) & 31, nt = (e >> 9) & 1, kt = e >> 10;
    int c = kt * 32 + kmap(ln >> 4, j);
    int tc = nt * 16 + (ln & 15);
    sBc[kt][nt][ln * 16 + j] = f2bf(sLc[c][tc]);
    sBp[kt][nt][ln * 16 + j] = f2bf(sLp[c][tc]);
  }
  __syncthreads();

  const int aSize = F * 64;
  const unsigned short* At0 = frag;
  const unsigned short* At1 = frag + aSize;
  const unsigned short* Ag0 = frag + 2 * aSize;
  const unsigned short* Ag1 = frag + 3 * aSize;
  const unsigned short* Sfr = frag + 4 * aSize;

  // GEMM1: t_pre / g_pre tiles (M=F rows, N=32 cols), K=64
  const int nTiles = (F >> 4) * 2;
  for (int tile = wv; tile < nTiles; tile += 8) {
    int rt = tile >> 1, nt = tile & 1;
    v8f accT = {0.f,0.f,0.f,0.f,0.f,0.f,0.f,0.f};
    v8f accG = {0.f,0.f,0.f,0.f,0.f,0.f,0.f,0.f};
    for (int kt = 0; kt < 2; ++kt) {
      v16bf bc = *(const v16bf*)&sBc[kt][nt][lane * 16];
      v16bf bp = *(const v16bf*)&sBp[kt][nt][lane * 16];
      int ao = ((rt * 2 + kt) * 32 + lane) * 16;
      v16bf at1 = *(const v16bf*)(At1 + ao);
      v16bf at0 = *(const v16bf*)(At0 + ao);
      v16bf ag1 = *(const v16bf*)(Ag1 + ao);
      v16bf ag0 = *(const v16bf*)(Ag0 + ao);
      accT = WMMA_BF16(at1, bc, accT);
      accT = WMMA_BF16(at0, bp, accT);
      accG = WMMA_BF16(ag1, bc, accG);
      accG = WMMA_BF16(ag0, bp, accG);
    }
    // epilogue: bias + global cond + local cond + gating, write m fragments
    int col = lane & 15;
    int tc = nt * 16 + col;
    float yp = sYp[tc], yc = sYc[tc];
    int mhi = (lane >> 4) << 3;
    for (int v = 0; v < 8; ++v) {
      int f = rt * 16 + v + mhi;
      float tv = accT[v] + dil_tb[f] + loc_tb[f] + gcT[b * F + f]
               + locTW[2 * f] * yp + locTW[2 * f + 1] * yc;
      float gv = accG[v] + dil_gb[f] + loc_gb[f] + gcG[b * F + f]
               + locGW[2 * f] * yp + locGW[2 * f + 1] * yc;
      float m = tanhf(tv) * (1.f / (1.f + __expf(-gv)));
      int k = f & 31, kt2 = f >> 5;
      int g2 = (k >> 3) & 1;
      int j2 = (k & 7) + ((k >= 16) ? 8 : 0);
      sM[kt2][nt][(g2 * 16 + col) * 16 + j2] = f2bf(m);
    }
  }
  __syncthreads();

  // GEMM2: s = scale_W @ m  (64 x 32), K=F ; then residual + skip update
  {
    int rt = wv >> 1, nt = wv & 1;     // 8 waves -> 4 row tiles x 2 col tiles
    v8f acc = {0.f,0.f,0.f,0.f,0.f,0.f,0.f,0.f};
    int kTiles = F >> 5;
    for (int kt = 0; kt < kTiles; ++kt) {
      v16bf a = *(const v16bf*)(Sfr + ((rt * kTiles + kt) * 32 + lane) * 16);
      v16bf bm = *(const v16bf*)&sM[kt][nt][lane * 16];
      acc = WMMA_BF16(a, bm, acc);
    }
    int col = lane & 15, tc = nt * 16 + col;
    int mhi = (lane >> 4) << 3;
    for (int v = 0; v < 8; ++v) {
      int o = rt * 16 + v + mhi;
      float s = acc[v] + scale_b[o];
      size_t idx = (size_t)b * 64 * T + (size_t)o * T + t0 + tc;
      l_out[idx] = l_in[idx] + s;
      skip[idx] += s;
    }
  }
}

// ---------------------------------------------------------------------------
// Fused head: relu(skip) -> pre(128) -> relu -> out(256) -> softmax(channels)
// ---------------------------------------------------------------------------
__global__ __launch_bounds__(256) void wavenet_head(
    const float* __restrict__ skip,
    const unsigned short* __restrict__ preF, const unsigned short* __restrict__ outF,
    const float* __restrict__ pre_b, const float* __restrict__ out_b,
    float* __restrict__ out, int T)
{
  __shared__ float sS[64][32];
  __shared__ __align__(32) unsigned short sR[2][2][512];  // relu(skip) frags K=64
  __shared__ __align__(32) unsigned short sP[4][2][512];  // relu(p) frags  K=128
  __shared__ float sO[256 * 32];
  __shared__ float sRed[8][32];
  __shared__ float sMax[32], sSum[32];

  const int b = blockIdx.y;
  const int t0 = blockIdx.x * 32;
  const int tid = threadIdx.x;
  const int lane = tid & 31, wv = tid >> 5;
  const float* sb = skip + (size_t)b * 64 * T;

  for (int e = tid; e < 2048; e += 256) {
    int c = e >> 5, tc = e & 31;
    sS[c][tc] = fmaxf(sb[(size_t)c * T + t0 + tc], 0.f);
  }
  __syncthreads();
  for (int e = tid; e < 2048; e += 256) {
    int j = e & 15, ln = (e >> 4) & 31, nt = (e >> 9) & 1, kt = e >> 10;
    int c = kt * 32 + kmap(ln >> 4, j);
    int tc = nt * 16 + (ln & 15);
    sR[kt][nt][ln * 16 + j] = f2bf(sS[c][tc]);
  }
  __syncthreads();

  // pre: 128 x 32, K=64 -> 16 tiles over 8 waves
  for (int tile = wv; tile < 16; tile += 8) {
    int rt = tile >> 1, nt = tile & 1;
    v8f acc = {0.f,0.f,0.f,0.f,0.f,0.f,0.f,0.f};
    for (int kt = 0; kt < 2; ++kt) {
      v16bf a = *(const v16bf*)(preF + ((rt * 2 + kt) * 32 + lane) * 16);
      v16bf bv = *(const v16bf*)&sR[kt][nt][lane * 16];
      acc = WMMA_BF16(a, bv, acc);
    }
    int col = lane & 15, mhi = (lane >> 4) << 3;
    for (int v = 0; v < 8; ++v) {
      int f = rt * 16 + v + mhi;
      float p = fmaxf(acc[v] + pre_b[f], 0.f);
      int k = f & 31, kt2 = f >> 5;
      int g2 = (k >> 3) & 1;
      int j2 = (k & 7) + ((k >= 16) ? 8 : 0);
      sP[kt2][nt][(g2 * 16 + col) * 16 + j2] = f2bf(p);
    }
  }
  __syncthreads();

  // out: 256 x 32, K=128 -> 32 tiles over 8 waves
  for (int tile = wv; tile < 32; tile += 8) {
    int rt = tile >> 1, nt = tile & 1;
    v8f acc = {0.f,0.f,0.f,0.f,0.f,0.f,0.f,0.f};
    for (int kt = 0; kt < 4; ++kt) {
      v16bf a = *(const v16bf*)(outF + ((rt * 4 + kt) * 32 + lane) * 16);
      v16bf bv = *(const v16bf*)&sP[kt][nt][lane * 16];
      acc = WMMA_BF16(a, bv, acc);
    }
    int col = lane & 15, tc = nt * 16 + col;
    int mhi = (lane >> 4) << 3;
    for (int v = 0; v < 8; ++v) {
      int o = rt * 16 + v + mhi;
      sO[o * 32 + tc] = acc[v] + out_b[o];
    }
  }
  __syncthreads();

  // softmax over 256 channels; 8 segments x 32 columns
  int col = tid & 31, seg = tid >> 5;
  float mx = -3.4e38f;
  for (int r = seg * 32; r < seg * 32 + 32; ++r) mx = fmaxf(mx, sO[r * 32 + col]);
  sRed[seg][col] = mx;
  __syncthreads();
  if (seg == 0) {
    float m = sRed[0][col];
    for (int s2 = 1; s2 < 8; ++s2) m = fmaxf(m, sRed[s2][col]);
    sMax[col] = m;
  }
  __syncthreads();
  float gm = sMax[col];
  float sum = 0.f;
  for (int r = seg * 32; r < seg * 32 + 32; ++r) sum += __expf(sO[r * 32 + col] - gm);
  sRed[seg][col] = sum;
  __syncthreads();
  if (seg == 0) {
    float s = 0.f;
    for (int s2 = 0; s2 < 8; ++s2) s += sRed[s2][col];
    sSum[col] = s;
  }
  __syncthreads();
  float inv = 1.f / sSum[col];
  for (int r = seg * 32; r < seg * 32 + 32; ++r)
    out[(size_t)b * 256 * T + (size_t)r * T + t0 + col] =
        __expf(sO[r * 32 + col] - gm) * inv;
}

// ---------------------------------------------------------------------------
extern "C" void kernel_launch(void* const* d_in, const int* in_sizes, int n_in,
                              void* d_out, int out_size, void* d_ws, size_t ws_size,
                              hipStream_t stream) {
  static const int FILT[9] = {96, 96, 96, 96, 96, 128, 128, 128, 128};
  const int T = in_sizes[0] / 8;  // data is [8,1,T]

  // d_in index map (setup_inputs dict order, lists flattened):
  // 0 data, 1 y, 2 h, 3 causal_W, 4 causal_b,
  // 5..13 dil_tanh_W, 14..22 dil_tanh_b, 23..31 dil_gate_W, 32..40 dil_gate_b,
  // 41..49 glob_tanh_W, 50..58 glob_gate_W, 59..67 loc_tanh_W, 68..76 loc_tanh_b,
  // 77..85 loc_gate_W, 86..94 loc_gate_b, 95..103 scale_W, 104..112 scale_b,
  // 113 pre_W, 114 pre_b, 115 out_W, 116 out_b
  const float* data = (const float*)d_in[0];
  const float* y    = (const float*)d_in[1];
  const float* h    = (const float*)d_in[2];
  const float* cW   = (const float*)d_in[3];
  const float* cb   = (const float*)d_in[4];

  char* ws = (char*)d_ws;
  const size_t LB = (size_t)8 * 64 * T * sizeof(float);
  float* l0   = (float*)(ws);
  float* l1   = (float*)(ws + LB);
  float* skip = (float*)(ws + 2 * LB);
  const size_t BLK = 262144;  // per-block packed-weight stride (bytes)
  char* wbase = ws + 3 * LB;
  unsigned short* preF = (unsigned short*)(wbase + 9 * BLK);
  unsigned short* outF = preF + 128 * 64;

  // pack per-block weights + global conditioning
  for (int i = 0; i < 9; ++i) {
    int F = FILT[i];
    unsigned short* frag = (unsigned short*)(wbase + i * BLK);
    float* gcT = (float*)(wbase + i * BLK + (size_t)640 * F);
    float* gcG = gcT + 8 * F;
    int total = 136 * F;
    hipLaunchKernelGGL(pack_block, dim3((total + 255) / 256), dim3(256), 0, stream,
                       (const float*)d_in[5 + i], (const float*)d_in[23 + i],
                       (const float*)d_in[95 + i], (const float*)d_in[41 + i],
                       (const float*)d_in[50 + i], h, frag, gcT, gcG, F);
  }
  hipLaunchKernelGGL(pack_head, dim3(160), dim3(256), 0, stream,
                     (const float*)d_in[113], (const float*)d_in[115], preF, outF);

  // initial causal conv + skip init
  long N = (long)8 * 64 * T;
  hipLaunchKernelGGL(wavenet_init, dim3((unsigned)((N + 255) / 256)), dim3(256), 0,
                     stream, data, cW, cb, l0, skip, T);

  // 9 fused residual blocks (ping-pong l buffers)
  dim3 grid(T / 32, 8);
  for (int i = 0; i < 9; ++i) {
    int F = FILT[i], d = 2 << i;
    const float* lin = (i & 1) ? l1 : l0;
    float* lout      = (i & 1) ? l0 : l1;
    unsigned short* frag = (unsigned short*)(wbase + i * BLK);
    float* gcT = (float*)(wbase + i * BLK + (size_t)640 * F);
    float* gcG = gcT + 8 * F;
    hipLaunchKernelGGL(wavenet_block, grid, dim3(256), 0, stream,
                       lin, lout, skip, y, frag, gcT, gcG,
                       (const float*)d_in[14 + i], (const float*)d_in[68 + i],
                       (const float*)d_in[32 + i], (const float*)d_in[86 + i],
                       (const float*)d_in[59 + i], (const float*)d_in[77 + i],
                       (const float*)d_in[104 + i], F, d, T);
  }

  // fused head: relu -> pre -> relu -> out -> softmax
  hipLaunchKernelGGL(wavenet_head, grid, dim3(256), 0, stream,
                     skip, preF, outF, (const float*)d_in[114],
                     (const float*)d_in[116], (float*)d_out, T);
}
